// LocalInference_Layer_89215060673299
// MI455X (gfx1250) — matrix operand order repaired and checked by reading, not compiled, not run on hardware
//
#include <hip/hip_runtime.h>

#define BB 64
#define LL 512
#define DD 600
#define DP 608

typedef __attribute__((ext_vector_type(16))) __bf16 v16bf;
typedef __attribute__((ext_vector_type(8)))  __bf16 v8bf;
typedef __attribute__((ext_vector_type(8)))  float  v8f;

__device__ __forceinline__ v8f wmma_bf16(v16bf a, v16bf b, v8f c) {
  return __builtin_amdgcn_wmma_f32_16x16x32_bf16(
      /*neg_a=*/false, a, /*neg_b=*/false, b,
      /*c_mod=*/(short)0, c, /*reuse_a=*/false, /*reuse_b=*/false);
}

// A-matrix 16x32 bf16 fragment (ISA 7.12.2): lane holds row m = l&15.
// half = l>>4: elem i<8 -> K = kk + 8*half + i ; elem i>=8 -> K = kk + 16 + 8*half + (i-8)
__device__ __forceinline__ v16bf load_a_bf(const __bf16* __restrict__ row, int kk, int half) {
  const v8bf lo = *(const v8bf*)(row + kk + 8 * half);
  const v8bf hi = *(const v8bf*)(row + kk + 16 + 8 * half);
  v16bf a;
#pragma unroll
  for (int i = 0; i < 8; ++i) { a[i] = lo[i]; a[8 + i] = hi[i]; }
  return a;
}

// B-matrix 32x16 bf16 fragment: lane holds col n = l&15; elem i<8 -> K = kk + 8*(l>>4) + i ; elem i>=8 -> K = kk + 16 + 8*(l>>4) + (i-8)
__device__ __forceinline__ v16bf load_b_bf(const __bf16* __restrict__ row, int kk, int kgrp) {
  const v8bf lo = *(const v8bf*)(row + kk + 8 * kgrp);
  const v8bf hi = *(const v8bf*)(row + kk + 16 + 8 * kgrp);
  v16bf b;
#pragma unroll
  for (int i = 0; i < 8; ++i) { b[i] = lo[i]; b[8 + i] = hi[i]; }
  return b;
}

// A fragment from f32 scores with fused softmax: exp(x - mx) * rinv, cast to bf16.
__device__ __forceinline__ v16bf load_a_sm(const float* __restrict__ row, int kk, int half,
                                           float mx, float rinv) {
  const float4 l0 = *(const float4*)(row + kk + 8 * half);
  const float4 l1 = *(const float4*)(row + kk + 8 * half + 4);
  const float4 h0 = *(const float4*)(row + kk + 16 + 8 * half);
  const float4 h1 = *(const float4*)(row + kk + 16 + 8 * half + 4);
  float v[16] = {l0.x, l0.y, l0.z, l0.w, l1.x, l1.y, l1.z, l1.w,
                 h0.x, h0.y, h0.z, h0.w, h1.x, h1.y, h1.z, h1.w};
  v16bf a;
#pragma unroll
  for (int i = 0; i < 16; ++i) a[i] = (__bf16)(__expf(v[i] - mx) * rinv);
  return a;
}

// ---------------------------------------------------------------------------
// Kernel 1: f32 [L][DD] -> bf16 [L][DP] (zero-padded) + bf16 transposed [DP][L]
// block (32,8), grid (L/32, DP/32, B)
// ---------------------------------------------------------------------------
__global__ void cvt_tr_kernel(const float* __restrict__ src, __bf16* __restrict__ dst,
                              __bf16* __restrict__ dstT) {
  __shared__ __bf16 tile[32][33];
  const int b  = blockIdx.z;
  const int i0 = blockIdx.x * 32;
  const int d0 = blockIdx.y * 32;
  const float* s = src + (size_t)b * LL * DD;
  __bf16* dd = dst + (size_t)b * LL * DP;
  __bf16* dt = dstT + (size_t)b * DP * LL;
  const int tx = threadIdx.x, ty = threadIdx.y;
#pragma unroll
  for (int r = 0; r < 4; ++r) {
    const int i = i0 + ty + r * 8;
    const int d = d0 + tx;
    const float v = (d < DD) ? s[(size_t)i * DD + d] : 0.0f;
    const __bf16 bv = (__bf16)v;
    dd[(size_t)i * DP + d] = bv;
    tile[ty + r * 8][tx] = bv;
  }
  __syncthreads();
#pragma unroll
  for (int r = 0; r < 4; ++r) {
    const int d = d0 + ty + r * 8;
    const int i = i0 + tx;
    dt[(size_t)d * LL + i] = tile[tx][ty + r * 8];
  }
}

// ---------------------------------------------------------------------------
// Kernel 2: e[b] = p_bf[b] (LLxDP) * h_bf[b]^T -> f32 e (LLxLL)
// block 128 (4 waves), each wave a 32x32 tile; grid (64, B)
// ---------------------------------------------------------------------------
__global__ void gemm_e_kernel(const __bf16* __restrict__ pbf, const __bf16* __restrict__ hbf,
                              float* __restrict__ e) {
  const int b    = blockIdx.y;
  const int wave = threadIdx.x >> 5;
  const int lane = threadIdx.x & 31;
  const int region = blockIdx.x * 4 + wave;     // 0..255
  const int mt = region >> 4, nt = region & 15;
  const int m0 = mt * 32, n0 = nt * 32;
  const int half = lane >> 4, l16 = lane & 15;

  const __bf16* arow0 = pbf + ((size_t)b * LL + (m0 + l16)) * DP;
  const __bf16* arow1 = arow0 + (size_t)16 * DP;
  const __bf16* brow0 = hbf + ((size_t)b * LL + (n0 + l16)) * DP;
  const __bf16* brow1 = brow0 + (size_t)16 * DP;

  v8f acc00 = {}, acc01 = {}, acc10 = {}, acc11 = {};
  for (int kk = 0; kk < DP; kk += 32) {
    __builtin_prefetch(arow0 + kk + 32, 0, 1);
    __builtin_prefetch(brow0 + kk + 32, 0, 1);
    const v16bf a0 = load_a_bf(arow0, kk, half);
    const v16bf a1 = load_a_bf(arow1, kk, half);
    const v16bf b0 = load_b_bf(brow0, kk, half);
    const v16bf b1 = load_b_bf(brow1, kk, half);
    acc00 = wmma_bf16(a0, b0, acc00);
    acc01 = wmma_bf16(a0, b1, acc01);
    acc10 = wmma_bf16(a1, b0, acc10);
    acc11 = wmma_bf16(a1, b1, acc11);
  }

  float* eb = e + (size_t)b * LL * LL;
#pragma unroll
  for (int r = 0; r < 8; ++r) {
    const int mA = m0 + r + 8 * half;
    const int mB = mA + 16;
    const int nA = n0 + l16, nB = nA + 16;
    eb[(size_t)mA * LL + nA] = acc00[r];
    eb[(size_t)mA * LL + nB] = acc01[r];
    eb[(size_t)mB * LL + nA] = acc10[r];
    eb[(size_t)mB * LL + nB] = acc11[r];
  }
}

// ---------------------------------------------------------------------------
// Kernel 3: 512x512 f32 transpose per batch; block (32,8), grid (256, B)
// ---------------------------------------------------------------------------
__global__ void tr512_kernel(const float* __restrict__ src, float* __restrict__ dst) {
  __shared__ float tile[32][33];
  const int b = blockIdx.y;
  const int t = blockIdx.x;
  const int i0 = (t >> 4) * 32, j0 = (t & 15) * 32;
  const float* s = src + (size_t)b * LL * LL;
  float* d = dst + (size_t)b * LL * LL;
  const int tx = threadIdx.x, ty = threadIdx.y;
#pragma unroll
  for (int r = 0; r < 4; ++r)
    tile[ty + r * 8][tx] = s[(size_t)(i0 + ty + r * 8) * LL + j0 + tx];
  __syncthreads();
#pragma unroll
  for (int r = 0; r < 4; ++r)
    d[(size_t)(j0 + ty + r * 8) * LL + i0 + tx] = tile[tx][ty + r * 8];
}

// ---------------------------------------------------------------------------
// Kernel 4: per-row max & sum(exp(x-max)) over 512 cols. One wave per row.
// block 256 (8 waves), grid (B*LL/8)
// ---------------------------------------------------------------------------
__global__ void rowstats_kernel(const float* __restrict__ e, float* __restrict__ omax,
                                float* __restrict__ osum) {
  const int wave = threadIdx.x >> 5, lane = threadIdx.x & 31;
  const int row = blockIdx.x * 8 + wave;        // 0 .. B*LL-1
  const float* r = e + (size_t)row * LL;
  float vals[16];
  float m = -3.0e38f;
#pragma unroll
  for (int t = 0; t < 16; ++t) { vals[t] = r[lane + t * 32]; m = fmaxf(m, vals[t]); }
#pragma unroll
  for (int o = 16; o > 0; o >>= 1) m = fmaxf(m, __shfl_xor(m, o, 32));
  float s = 0.0f;
#pragma unroll
  for (int t = 0; t < 16; ++t) s += __expf(vals[t] - m);
#pragma unroll
  for (int o = 16; o > 0; o >>= 1) s += __shfl_xor(s, o, 32);
  if (lane == 0) { omax[row] = m; osum[row] = s; }
}

// ---------------------------------------------------------------------------
// Kernel 5: out[m, 0:4D] = [x, x_, x-x_, x*x_] with x_ = softmax(esc) * Bmat.
// esc: f32 [LL][LL] per batch (row softmax via smax/ssum), Bmat: bf16 [DP][LL]
// (transposed so K is contiguous), src: f32 [LL][DD].
// block 128 (4 waves), wave = 32x32 tile; grid (76, B)  (16 M-tiles x 19 N-tiles)
// ---------------------------------------------------------------------------
__device__ __forceinline__ void store4(float* __restrict__ out, const float* __restrict__ src,
                                       int b, int m, int n, float att) {
  if (n >= DD) return;
  const float pv = src[((size_t)b * LL + m) * DD + n];
  float* o = out + ((size_t)b * LL + m) * (4 * DD);
  o[n]          = pv;
  o[DD + n]     = att;
  o[2 * DD + n] = pv - att;
  o[3 * DD + n] = pv * att;
}

__global__ void attend_kernel(const float* __restrict__ esc, const float* __restrict__ smax,
                              const float* __restrict__ ssum, const __bf16* __restrict__ bt,
                              const float* __restrict__ src, float* __restrict__ out) {
  const int b    = blockIdx.y;
  const int wave = threadIdx.x >> 5;
  const int lane = threadIdx.x & 31;
  const int region = blockIdx.x * 4 + wave;
  if (region >= 16 * 19) return;
  const int mt = region / 19, nt = region % 19;
  const int m0 = mt * 32, n0 = nt * 32;
  const int half = lane >> 4, l16 = lane & 15;

  const int mA = m0 + l16, mB = mA + 16;
  const float* ea0 = esc + ((size_t)b * LL + mA) * LL;
  const float* ea1 = esc + ((size_t)b * LL + mB) * LL;
  const float mx0 = smax[b * LL + mA], ri0 = 1.0f / ssum[b * LL + mA];
  const float mx1 = smax[b * LL + mB], ri1 = 1.0f / ssum[b * LL + mB];
  const __bf16* bb0 = bt + ((size_t)b * DP + (n0 + l16)) * LL;
  const __bf16* bb1 = bb0 + (size_t)16 * LL;

  v8f acc00 = {}, acc01 = {}, acc10 = {}, acc11 = {};
  for (int kk = 0; kk < LL; kk += 32) {
    __builtin_prefetch(ea0 + kk + 32, 0, 1);
    __builtin_prefetch(bb0 + kk + 32, 0, 1);
    const v16bf a0 = load_a_sm(ea0, kk, half, mx0, ri0);
    const v16bf a1 = load_a_sm(ea1, kk, half, mx1, ri1);
    const v16bf b0 = load_b_bf(bb0, kk, half);
    const v16bf b1 = load_b_bf(bb1, kk, half);
    acc00 = wmma_bf16(a0, b0, acc00);
    acc01 = wmma_bf16(a0, b1, acc01);
    acc10 = wmma_bf16(a1, b0, acc10);
    acc11 = wmma_bf16(a1, b1, acc11);
  }

#pragma unroll
  for (int r = 0; r < 8; ++r) {
    const int mlo = m0 + r + 8 * half;
    const int mhi = mlo + 16;
    const int nA = n0 + l16, nB = nA + 16;
    store4(out, src, b, mlo, nA, acc00[r]);
    store4(out, src, b, mlo, nB, acc01[r]);
    store4(out, src, b, mhi, nA, acc10[r]);
    store4(out, src, b, mhi, nB, acc11[r]);
  }
}

// ---------------------------------------------------------------------------
extern "C" void kernel_launch(void* const* d_in, const int* in_sizes, int n_in,
                              void* d_out, int out_size, void* d_ws, size_t ws_size,
                              hipStream_t stream) {
  (void)in_sizes; (void)n_in; (void)out_size; (void)ws_size;
  const float* p = (const float*)d_in[0];
  const float* h = (const float*)d_in[1];

  // workspace layout
  const size_t szBF = (size_t)BB * LL * DP;        // bf16 elements per buffer
  __bf16* pbf = (__bf16*)d_ws;
  __bf16* hbf = pbf + szBF;
  __bf16* pT  = hbf + szBF;
  __bf16* hT  = pT + szBF;
  float* e    = (float*)(hT + szBF);
  const size_t EL = (size_t)BB * LL * LL;
  float* eT   = e + EL;
  float* rmax = eT + EL;
  float* rsum = rmax + (size_t)BB * LL;
  float* cmax = rsum + (size_t)BB * LL;
  float* csum = cmax + (size_t)BB * LL;

  float* outP = (float*)d_out;
  float* outH = outP + (size_t)BB * LL * (4 * DD);

  dim3 cvtBlk(32, 8), cvtGrd(LL / 32, DP / 32, BB);
  cvt_tr_kernel<<<cvtGrd, cvtBlk, 0, stream>>>(p, pbf, pT);
  cvt_tr_kernel<<<cvtGrd, cvtBlk, 0, stream>>>(h, hbf, hT);

  gemm_e_kernel<<<dim3(64, BB), 128, 0, stream>>>(pbf, hbf, e);

  tr512_kernel<<<dim3(256, BB), dim3(32, 8), 0, stream>>>(e, eT);

  rowstats_kernel<<<dim3(BB * LL / 8), 256, 0, stream>>>(e, rmax, rsum);
  rowstats_kernel<<<dim3(BB * LL / 8), 256, 0, stream>>>(eT, cmax, csum);

  // p_ = rowsoftmax(e) * h   -> m_p
  attend_kernel<<<dim3(76, BB), 128, 0, stream>>>(e, rmax, rsum, hT, p, outP);
  // h_ = colsoftmax(e)^T * p -> m_h   (colsoftmax of e == rowsoftmax of eT)
  attend_kernel<<<dim3(76, BB), 128, 0, stream>>>(eT, cmax, csum, pT, h, outH);
}
